// Attention_56951266345146
// MI455X (gfx1250) — matrix operand order repaired and checked by reading, not compiled, not run on hardware
//
#include <hip/hip_runtime.h>

typedef _Float16 f16;
typedef _Float16 v8h  __attribute__((ext_vector_type(8)));
typedef _Float16 v16h __attribute__((ext_vector_type(16)));
typedef float    v8f  __attribute__((ext_vector_type(8)));

#define NIMG 10
#define CCH  640
#define HW   32
#define NPIX 1024
#define PADW 34
#define NH   8
#define HD   80
#define HDP  96
#define LOG2E 1.4426950408889634f
#define ATTN_SCALE 0.11180339887498949f   // 80^-0.5

__device__ __forceinline__ v16h cat16(v8h lo, v8h hi) {
  v16h r;
#pragma unroll
  for (int i = 0; i < 8; ++i) { r[i] = lo[i]; r[i + 8] = hi[i]; }
  return r;
}
__device__ __forceinline__ v8f zero8() {
  v8f z;
#pragma unroll
  for (int i = 0; i < 8; ++i) z[i] = 0.0f;
  return z;
}
__device__ __forceinline__ v8f wmma16(v16h a, v16h b, v8f c) {
  // D(16x16 f32) = A(16x32 f16) * B(32x16 f16) + C
  return __builtin_amdgcn_wmma_f32_16x16x32_f16(false, a, false, b, (short)0, c,
                                                false, false);
}

// ---------------------------------------------------------------------------
// Pack kernels (f32 -> f16, layout change)
// ---------------------------------------------------------------------------

// zero an f16 buffer (n multiple of 8)
__global__ __launch_bounds__(256) void zero_f16(f16* __restrict__ p, long n) {
  long i = ((long)blockIdx.x * blockDim.x + threadIdx.x) * 8;
  if (i < n) {
    v8h z;
#pragma unroll
    for (int j = 0; j < 8; ++j) z[j] = (f16)0.0f;
    *(v8h*)(p + i) = z;
  }
}

// x (10,640,32,32) f32 -> xpad (10,640,34,34) f16 with zero halo
__global__ __launch_bounds__(256) void pack_x(const float* __restrict__ x,
                                              f16* __restrict__ xpad) {
  int idx = blockIdx.x * 256 + threadIdx.x;
  const int TOT = NIMG * CCH * PADW * PADW;
  if (idx >= TOT) return;
  int gc = idx / (PADW * PADW);
  int pp = idx - gc * (PADW * PADW);
  int py = pp / PADW, px = pp - py * PADW;
  float v = 0.0f;
  if (py >= 1 && py <= HW && px >= 1 && px <= HW)
    v = x[(long)gc * NPIX + (py - 1) * HW + (px - 1)];
  xpad[idx] = (f16)v;
}

// wq/wk/wv (640,640,3,3) f32 -> wh (3,9,640,640) f16  (tap-major, ci contiguous)
__global__ __launch_bounds__(256) void pack_w(const float* __restrict__ wq,
                                              const float* __restrict__ wk,
                                              const float* __restrict__ wv,
                                              f16* __restrict__ wh) {
  long idx = (long)blockIdx.x * 256 + threadIdx.x;
  const long TOT = 3L * 9 * CCH * CCH;
  if (idx >= TOT) return;
  int proj = (int)(idx / (9L * CCH * CCH));
  long r   = idx - (long)proj * 9 * CCH * CCH;
  int tap  = (int)(r / ((long)CCH * CCH));
  long r2  = r - (long)tap * CCH * CCH;
  int co = (int)(r2 / CCH);
  int ci = (int)(r2 - (long)co * CCH);
  const float* w = (proj == 0) ? wq : (proj == 1 ? wk : wv);
  wh[idx] = (f16)w[((long)co * CCH + ci) * 9 + tap];
}

// w_proj (640,640) f32 -> f16 same layout
__global__ __launch_bounds__(256) void pack_wp(const float* __restrict__ wp,
                                               f16* __restrict__ wph) {
  int idx = blockIdx.x * 256 + threadIdx.x;
  if (idx < CCH * CCH) wph[idx] = (f16)wp[idx];
}

// ---------------------------------------------------------------------------
// Conv3x3 as implicit GEMM:  out[co][pix] = sum_{tap,ci} w[co][ci] * x[ci][pix']
// Block = 128 thr (4 waves). Block tile: 64 co x 64 pix (2 image rows).
// Wave w -> co tile [co_base, co_base+16), all 64 pixels (4 N-tiles).
// LDS stages 4 rows x 34 cols x 32 ci per K-chunk; reused by all 9 taps.
// Q,K stored as [g][h][pix][96] f16 (d contiguous); V stored transposed
// [g][h][d][pix] f16 (pix contiguous).
// ---------------------------------------------------------------------------
__global__ __launch_bounds__(128) void conv_qkv(
    const f16* __restrict__ xpad, const f16* __restrict__ wh,
    const float* __restrict__ bq, const float* __restrict__ bk,
    const float* __restrict__ bv, f16* __restrict__ qout,
    f16* __restrict__ kout, f16* __restrict__ vT) {
  __shared__ f16 lx[4][PADW][32];   // [row][col][ci]  8704 B

  int bid = blockIdx.x;
  int pt  = bid & 15;  bid >>= 4;     // pixel tile (64 pix)
  int cot = bid % 10;  bid /= 10;     // co tile (64 co)
  int g   = bid % 10;
  int proj = bid / 10;                // 0=q 1=k 2=v

  int lane = threadIdx.x & 31;
  int wave = threadIdx.x >> 5;
  int hi16 = lane >> 4;               // 0 or 1
  int l15  = lane & 15;

  int co_base = cot * 64 + wave * 16;
  int p_base  = pt * 64;
  int y0      = pt * 2;               // first image row of tile

  v8f acc[4];
#pragma unroll
  for (int i = 0; i < 4; ++i) acc[i] = zero8();

  int sci = threadIdx.x & 31;         // staging: ci
  int srr = threadIdx.x >> 5;         // staging: row 0..3

  // staging source for chunk 0 (advances by 32 channel planes per chunk)
  const f16* src =
      xpad + ((long)(g * CCH + sci) * PADW + (y0 + srr)) * PADW;
  const long CHUNK_STRIDE = 32L * PADW * PADW;
  // weight row base for this wave's co tile (chunk 0, tap 0)
  const f16* wbase = wh + (((long)proj * 9 * CCH + co_base + l15) * CCH) + hi16 * 8;
  const long WTAP_STRIDE = (long)CCH * CCH;   // per-tap stride in halves

  for (int cib = 0; cib < 20; ++cib) {
    __syncthreads();
    // stage xpad rows y0..y0+3, cols 0..33 (padded), 32 ci
    {
#pragma unroll 2
      for (int cc = 0; cc < PADW; ++cc) lx[srr][cc][sci] = src[cc];
    }
    // branch-free prefetch of next chunk (speculative; OOB tails are dropped)
    __builtin_prefetch(src + CHUNK_STRIDE, 0, 1);          // global_prefetch_b8
    __builtin_prefetch(src + CHUNK_STRIDE + PADW, 0, 1);
#pragma unroll
    for (int tap = 0; tap < 9; ++tap)
      __builtin_prefetch(wbase + tap * WTAP_STRIDE + (cib + 1) * 32, 0, 1);
    src += CHUNK_STRIDE;
    __syncthreads();

#pragma unroll
    for (int tap = 0; tap < 9; ++tap) {
      int ky = tap / 3, kx = tap - ky * 3;
      // A fragment: w[co_base + m][cib*32 + k], tap-major layout
      const f16* wrow = wbase + tap * WTAP_STRIDE + cib * 32;
      v16h afrag = cat16(*(const v8h*)wrow, *(const v8h*)(wrow + 16));
#pragma unroll
      for (int nt = 0; nt < 4; ++nt) {
        int p_local = nt * 16 + l15;
        int py = p_local >> 5, px = p_local & 31;
        const f16* bsrc = &lx[py + ky][px + kx][hi16 * 8];
        v16h bfrag = cat16(*(const v8h*)bsrc, *(const v8h*)(bsrc + 16));
        acc[nt] = wmma16(afrag, bfrag, acc[nt]);
      }
    }
  }

  // epilogue: bias + store
  const float* bias = (proj == 0) ? bq : (proj == 1 ? bk : bv);
  const float* bp = bias + co_base + hi16 * 8;
  int h  = co_base / HD;                       // co tiles never cross heads
  int d0 = co_base - h * HD + hi16 * 8;
  if (proj < 2) {
    f16* dst_base = (proj == 0) ? qout : kout;
#pragma unroll
    for (int nt = 0; nt < 4; ++nt) {
      int pix = p_base + nt * 16 + l15;
      v8h o;
#pragma unroll
      for (int r = 0; r < 8; ++r) o[r] = (f16)(acc[nt][r] + bp[r]);
      *(v8h*)(dst_base + ((long)(g * NH + h) * NPIX + pix) * HDP + d0) = o;
    }
  } else {
#pragma unroll
    for (int nt = 0; nt < 4; ++nt) {
      int pix = p_base + nt * 16 + l15;
#pragma unroll
      for (int r = 0; r < 8; ++r)
        vT[((long)(g * NH + h) * HD + d0 + r) * NPIX + pix] =
            (f16)(acc[nt][r] + bp[r]);
    }
  }
}

// ---------------------------------------------------------------------------
// Flash attention: one wave handles 16 queries of one (g,h); streams 1024 keys
// in chunks of 32 with online softmax. oh[(b*1024+pix)*5+m][h*80+d] f16.
// ---------------------------------------------------------------------------
__global__ __launch_bounds__(128) void attn_flash(
    const f16* __restrict__ q, const f16* __restrict__ k,
    const f16* __restrict__ vT, f16* __restrict__ oh) {
  __shared__ f16 pbuf[4][16][32];    // per-wave P transpose bounce, 4 KB

  int lane = threadIdx.x & 31;
  int w    = threadIdx.x >> 5;
  int hi16 = lane >> 4;
  int l15  = lane & 15;

  int wid   = blockIdx.x * 4 + w;
  int group = wid >> 6;              // 0..79 = g*8 + h
  int qt    = wid & 63;
  int g = group >> 3, h = group & 7;
  int b = g / 5, m = g - b * 5;

  // Q A-fragments (3 K-chunks over padded hd=96)
  const f16* qrow = q + ((long)group * NPIX + qt * 16 + l15) * HDP;
  v16h qa[3];
#pragma unroll
  for (int ck = 0; ck < 3; ++ck) {
    const f16* p = qrow + ck * 32 + hi16 * 8;
    qa[ck] = cat16(*(const v8h*)p, *(const v8h*)(p + 16));
  }

  v8f o[5];
#pragma unroll
  for (int i = 0; i < 5; ++i) o[i] = zero8();
  float mr[8], lr[8];
#pragma unroll
  for (int r = 0; r < 8; ++r) { mr[r] = -3.0e38f; lr[r] = 0.0f; }

  const f16* kbase = k + (long)group * NPIX * HDP;
  const f16* vbase = vT + (long)group * HD * NPIX;

  for (int kc = 0; kc < 32; ++kc) {
    // branch-free prefetch of next key chunk (speculative; OOB tail dropped)
    __builtin_prefetch(kbase + (long)((kc + 1) * 32 + l15) * HDP, 0, 1);
    __builtin_prefetch(kbase + (long)((kc + 1) * 32 + 16 + l15) * HDP, 0, 1);
    __builtin_prefetch(vbase + (long)l15 * NPIX + (kc + 1) * 32, 0, 1);

    // S = scale * Q K^T for 32 keys (two 16-col tiles)
    v8f s0 = zero8(), s1 = zero8();
#pragma unroll
    for (int ck = 0; ck < 3; ++ck) {
      const f16* kp0 = kbase + (long)(kc * 32 + l15) * HDP + ck * 32 + hi16 * 8;
      s0 = wmma16(qa[ck], cat16(*(const v8h*)kp0, *(const v8h*)(kp0 + 16)), s0);
      const f16* kp1 = kp0 + 16 * HDP;
      s1 = wmma16(qa[ck], cat16(*(const v8h*)kp1, *(const v8h*)(kp1 + 16)), s1);
    }
#pragma unroll
    for (int i = 0; i < 8; ++i) { s0[i] *= ATTN_SCALE; s1[i] *= ATTN_SCALE; }

    // online softmax (rows r + 8*hi16; 16 columns live in 16-lane halves)
    float corr[8];
#pragma unroll
    for (int r = 0; r < 8; ++r) {
      float v = fmaxf(s0[r], s1[r]);
#pragma unroll
      for (int msk = 1; msk < 16; msk <<= 1) v = fmaxf(v, __shfl_xor(v, msk, 32));
      float mnew = fmaxf(mr[r], v);
      corr[r] = exp2f((mr[r] - mnew) * LOG2E);
      mr[r] = mnew;
      float p0 = exp2f((s0[r] - mnew) * LOG2E);
      float p1 = exp2f((s1[r] - mnew) * LOG2E);
      s0[r] = p0; s1[r] = p1;
      float t = p0 + p1;
#pragma unroll
      for (int msk = 1; msk < 16; msk <<= 1) t += __shfl_xor(t, msk, 32);
      lr[r] = lr[r] * corr[r] + t;
    }
#pragma unroll
    for (int nt = 0; nt < 5; ++nt)
#pragma unroll
      for (int r = 0; r < 8; ++r) o[nt][r] *= corr[r];

    // transpose P (D-layout) into A-layout via per-wave LDS
    int m0 = hi16 * 8;
#pragma unroll
    for (int r = 0; r < 8; ++r) {
      pbuf[w][m0 + r][l15]      = (f16)s0[r];
      pbuf[w][m0 + r][16 + l15] = (f16)s1[r];
    }
    const f16* pp = &pbuf[w][l15][hi16 * 8];
    v16h pfrag = cat16(*(const v8h*)pp, *(const v8h*)(pp + 16));

    // O += P * V
#pragma unroll
    for (int nt = 0; nt < 5; ++nt) {
      const f16* vp = vbase + (long)(nt * 16 + l15) * NPIX + kc * 32 + hi16 * 8;
      o[nt] = wmma16(pfrag, cat16(*(const v8h*)vp, *(const v8h*)(vp + 16)), o[nt]);
    }
  }

  // normalize + store to oh[(b*1024+pix)*5+m][h*80+d]
#pragma unroll
  for (int r = 0; r < 8; ++r) {
    float inv = 1.0f / lr[r];
    int pix = qt * 16 + r + hi16 * 8;
    long row = ((long)b * NPIX + pix) * 5 + m;
#pragma unroll
    for (int nt = 0; nt < 5; ++nt)
      oh[row * CCH + h * HD + nt * 16 + l15] = (f16)(o[nt][r] * inv);
  }
}

// ---------------------------------------------------------------------------
// Output projection: out[row][d] = sum_c oh[row][c] * w_proj[d][c] + b[d]
// rows = (b*1024+n)*5+m = 10240. Block 128 thr; wave w: 16 rows x 64 dout.
// ---------------------------------------------------------------------------
__global__ __launch_bounds__(128) void proj_gemm(
    const f16* __restrict__ oh, const f16* __restrict__ wph,
    const float* __restrict__ bproj, float* __restrict__ out) {
  int lane = threadIdx.x & 31;
  int w    = threadIdx.x >> 5;
  int hi16 = lane >> 4;
  int l15  = lane & 15;

  int rt = blockIdx.x / 10;
  int dt = blockIdx.x - rt * 10;
  int rbase = rt * 64 + w * 16;
  int dbase = dt * 64;

  v8f acc[4];
#pragma unroll
  for (int i = 0; i < 4; ++i) acc[i] = zero8();

  for (int cb = 0; cb < 20; ++cb) {
    const f16* ap = oh + (long)(rbase + l15) * CCH + cb * 32 + hi16 * 8;
    v16h afrag = cat16(*(const v8h*)ap, *(const v8h*)(ap + 16));
    __builtin_prefetch(ap + 32, 0, 1);   // branch-free next-chunk prefetch
#pragma unroll
    for (int j = 0; j < 4; ++j) {
      const f16* bpp = wph + (long)(dbase + j * 16 + l15) * CCH + cb * 32 + hi16 * 8;
      v16h bfrag = cat16(*(const v8h*)bpp, *(const v8h*)(bpp + 16));
      acc[j] = wmma16(afrag, bfrag, acc[j]);
    }
  }

#pragma unroll
  for (int j = 0; j < 4; ++j) {
    int d = dbase + j * 16 + l15;
    float bb = bproj[d];
#pragma unroll
    for (int r = 0; r < 8; ++r) {
      long row = rbase + r + hi16 * 8;
      out[row * CCH + d] = acc[j][r] + bb;
    }
  }
}

// ---------------------------------------------------------------------------
extern "C" void kernel_launch(void* const* d_in, const int* in_sizes, int n_in,
                              void* d_out, int out_size, void* d_ws,
                              size_t ws_size, hipStream_t stream) {
  const float* x  = (const float*)d_in[0];
  const float* wq = (const float*)d_in[1];
  const float* bq = (const float*)d_in[2];
  const float* wk = (const float*)d_in[3];
  const float* bk = (const float*)d_in[4];
  const float* wv = (const float*)d_in[5];
  const float* bv = (const float*)d_in[6];
  const float* wp = (const float*)d_in[7];
  const float* bp = (const float*)d_in[8];
  float* out = (float*)d_out;

  char* base = (char*)d_ws;
  size_t off = 0;
  auto take = [&](size_t nbytes) -> void* {
    void* p = base + off;
    off += (nbytes + 255) & ~(size_t)255;
    return p;
  };

  const long XPAD_N = (long)NIMG * CCH * PADW * PADW;      // 7,398,400
  const long WH_N   = 3L * 9 * CCH * CCH;                  // 11,059,200
  const long QK_N   = (long)NIMG * NH * NPIX * HDP;        // 7,864,320
  const long VT_N   = (long)NIMG * NH * HD * NPIX;         // 6,553,600
  const long OH_N   = 2L * NPIX * 5 * CCH;                 // 6,553,600

  f16* xpad = (f16*)take(XPAD_N * 2);
  f16* wh   = (f16*)take(WH_N * 2);
  f16* wph  = (f16*)take((long)CCH * CCH * 2);
  f16* qh   = (f16*)take(QK_N * 2);
  f16* kh   = (f16*)take(QK_N * 2);
  f16* vT   = (f16*)take(VT_N * 2);
  f16* ohb  = (f16*)take(OH_N * 2);
  (void)ws_size;

  // zero q/k (hd padding 80..95 must be zero for the QK^T K-chunks)
  zero_f16<<<(int)(QK_N / 2048), 256, 0, stream>>>(qh, QK_N);
  zero_f16<<<(int)(QK_N / 2048), 256, 0, stream>>>(kh, QK_N);

  pack_x<<<(int)((XPAD_N + 255) / 256), 256, 0, stream>>>(x, xpad);
  pack_w<<<(int)((WH_N + 255) / 256), 256, 0, stream>>>(wq, wk, wv, wh);
  pack_wp<<<(CCH * CCH + 255) / 256, 256, 0, stream>>>(wp, wph);

  // conv: 3 proj x 10 img x 10 co-tiles x 16 pix-tiles
  conv_qkv<<<3 * 10 * 10 * 16, 128, 0, stream>>>(xpad, wh, bq, bk, bv, qh, kh, vT);

  // attention: 80 (g,h) groups x 64 query tiles, 4 waves/block
  attn_flash<<<80 * 64 / 4, 128, 0, stream>>>(qh, kh, vT, ohb);

  // projection: 160 row-tiles x 10 dout-tiles
  proj_gemm<<<160 * 10, 128, 0, stream>>>(ohb, wph, bp, out);
}
